// se3_model_11398843204128
// MI455X (gfx1250) — compile-verified, matrix operand order checked by reading, not run on hardware
//
#include <hip/hip_runtime.h>
#include <hip/hip_bf16.h>
#include <cstdint>
#include <cstddef>

// ---------------------------------------------------------------------------
// Types / constants
// ---------------------------------------------------------------------------
typedef __attribute__((ext_vector_type(16))) _Float16 v16h;
typedef __attribute__((ext_vector_type(8)))  float    v8f;

#define LL      512
#define NPOS    (512 * 512)
#define EDGES   16384

union AFrag {
    v16h     v;
    _Float16 h[16];
    unsigned u[8];
    uint4    x[2];
};

__device__ __forceinline__ float eluf(float x) {
    return x > 0.f ? x : (__expf(x) - 1.f);
}

// K-pair base for WMMA 16-bit A/B fragment register r, lane-half hi.
// reg 0..3 -> K = hi*8 + 2r ; reg 4..7 -> K = 16 + hi*8 + 2(r-4)
__device__ __forceinline__ int kbase_of(int r, int hi) {
    return ((r & 4) << 2) + hi * 8 + ((r & 3) << 1);
}

__device__ __forceinline__ v8f wmma16(v16h a, v16h b, v8f c) {
    // D = A(16x32 f16) * B(32x16 f16) + C(16x16 f32)
    return __builtin_amdgcn_wmma_f32_16x16x32_f16(
        /*neg_a=*/false, a, /*neg_b=*/false, b,
        /*c_mod=*/(short)0, c, /*reuse_a=*/false, /*reuse_b=*/false);
}

// CDNA5 async global->LDS copy (ASYNCcnt-tracked), 16B per active lane.
__device__ __forceinline__ void g2l_async_b128(void* lds_dst, const void* gsrc) {
    unsigned           l = (unsigned)(uintptr_t)lds_dst;   // LDS byte offset
    unsigned long long g = (unsigned long long)(uintptr_t)gsrc;
    asm volatile("global_load_async_to_lds_b128 %0, %1, off"
                 :
                 : "v"(l), "v"(g)
                 : "memory");
}
__device__ __forceinline__ void wait_async() {
    asm volatile("s_wait_asynccnt 0x0" ::: "memory");
}

// ---------------------------------------------------------------------------
// Small helpers
// ---------------------------------------------------------------------------
__global__ void k_zero(float* __restrict__ p, int n) {
    int i = blockIdx.x * blockDim.x + threadIdx.x;
    if (i < n) p[i] = 0.f;
}

__global__ void k_elu16(const _Float16* __restrict__ in, _Float16* __restrict__ out,
                        size_t n) {
    for (size_t i = (size_t)blockIdx.x * blockDim.x + threadIdx.x; i < n;
         i += (size_t)gridDim.x * blockDim.x)
        out[i] = (_Float16)eluf((float)in[i]);
}

// h1[c,l] = elu(sum_d c1_w[c,d] * f1d[d,l])      (16 x 512)
__global__ void k_h1(const float* __restrict__ f1d, const float* __restrict__ w,
                     const float* __restrict__ b, float* __restrict__ h1) {
    int j = blockIdx.x * blockDim.x + threadIdx.x;
    if (j >= LL) return;
    for (int c = 0; c < 16; ++c) {
        float s = b[c];
        for (int d = 0; d < 23; ++d) s += w[c * 23 + d] * f1d[d * LL + j];
        h1[c * LL + j] = eluf(s);
    }
}

// Pack OIHW f32 weights into *fragment-major* f16-pair dwords:
//   wpk[ ((block*NT + q)*32 + lane)*8 + r ]
// holds input-channel pair (chunk*32 + kbase(r,hi), +1) for out-channel
// n = q*16 + (lane&15).  block = tap*kchunks + chunk.
// Each lane's B fragment is then 32 contiguous bytes in LDS (2 x b128).
__global__ void k_prep(const float* __restrict__ w, unsigned* __restrict__ wpk,
                       int cout, int cin, int taps, int kchunks, int nt) {
    int total = taps * kchunks * nt * 256;
    for (int i = blockIdx.x * blockDim.x + threadIdx.x; i < total;
         i += gridDim.x * blockDim.x) {
        int r    = i & 7;
        int lane = (i >> 3) & 31;
        int bq   = i >> 8;
        int q    = bq % nt;
        int b    = bq / nt;
        int tap  = b / kchunks;
        int chunk = b % kchunks;
        int n  = q * 16 + (lane & 15);
        int hi = lane >> 4;
        int kb = kbase_of(r, hi);
        int c0 = chunk * 32 + kb, c1 = c0 + 1;
        float a  = (n < cout && c0 < cin) ? w[((size_t)n * cin + c0) * taps + tap] : 0.f;
        float bb = (n < cout && c1 < cin) ? w[((size_t)n * cin + c1) * taps + tap] : 0.f;
        union { _Float16 h[2]; unsigned u; } p;
        p.h[0] = (_Float16)a;
        p.h[1] = (_Float16)bb;
        wpk[i] = p.u;
    }
}

// ---------------------------------------------------------------------------
// Generic implicit-GEMM conv, f16 NHWC (Cpad=32), WMMA f32_16x16x32_f16.
// EPI: 0=store f16 (bias only), 1=bias+ELU f16, 2=bias+residual+ELU f16,
//      3=bias, store f32 (first `cout` channels) to out32 (stride 9).
// NT: number of 16-wide N tiles (2 -> 32 out channels, 1 -> <=16).
// ---------------------------------------------------------------------------
template <int EPI, int NT>
__global__ __launch_bounds__(128) void conv_wmma(
    const _Float16* __restrict__ in, const unsigned* __restrict__ wpk,
    const float* __restrict__ bias, _Float16* __restrict__ out16,
    float* __restrict__ out32, const _Float16* __restrict__ res, int taps,
    int dil, int cout) {
    __shared__ unsigned wlds[9 * 2 * 256];   // fragment-major weights
    int nw = taps * NT * 256;
    for (int i = threadIdx.x * 4; i < nw; i += 128 * 4)
        g2l_async_b128(&wlds[i], &wpk[i]);
    wait_async();
    __syncthreads();

    int wave = threadIdx.x >> 5, lane = threadIdx.x & 31;
    int m = lane & 15, hi = lane >> 4;
    const uint4 z4 = make_uint4(0u, 0u, 0u, 0u);

    for (int ti = 0; ti < 4; ++ti) {
        int tile = (blockIdx.x * 4 + wave) * 4 + ti;   // 16384 tiles total
        int y  = tile >> 5;
        int x0 = (tile & 31) << 4;

        v8f acc[NT];
#pragma unroll
        for (int q = 0; q < NT; ++q)
#pragma unroll
            for (int r = 0; r < 8; ++r) acc[q][r] = 0.f;

        // hint the tile's A patch toward the caches (global_prefetch_b8)
        __builtin_prefetch((const void*)(in + ((size_t)(y * LL + x0) << 5)), 0, 1);

        for (int t = 0; t < taps; ++t) {
            int dy = 0, dx = 0;
            if (taps == 9) {
                dy = ((t / 3) - 1) * dil;
                dx = ((t % 3) - 1) * dil;
            }
            int y2 = y + dy, x2 = x0 + m + dx;
            bool ok = ((unsigned)y2 < (unsigned)LL) & ((unsigned)x2 < (unsigned)LL);
            const uint4* ain4 =
                (const uint4*)(in + ((size_t)(y2 * LL + x2) << 5));
            AFrag A;
            A.x[0] = ok ? ain4[hi] : z4;       // regs 0..3  (global_load_b128)
            A.x[1] = ok ? ain4[2 + hi] : z4;   // regs 4..7
#pragma unroll
            for (int q = 0; q < NT; ++q) {
                AFrag B;
                const uint4* wq =
                    (const uint4*)&wlds[((t * NT + q) * 32 + lane) * 8];
                B.x[0] = wq[0];                // ds_load_b128
                B.x[1] = wq[1];
                acc[q] = wmma16(A.v, B.v, acc[q]);
            }
        }

        // Epilogue: lane holds (M = hi*8 + r, N = q*16 + (lane&15))
#pragma unroll
        for (int q = 0; q < NT; ++q) {
            int n = q * 16 + m;
            float bv;
            if constexpr (EPI == 3)
                bv = (n < cout) ? bias[n] : 0.f;
            else
                bv = bias[n];
#pragma unroll
            for (int r = 0; r < 8; ++r) {
                int mm = hi * 8 + r;
                size_t pos = (size_t)y * LL + x0 + mm;
                float v = acc[q][r] + bv;
                if constexpr (EPI == 0) {
                    out16[pos * 32 + n] = (_Float16)v;
                } else if constexpr (EPI == 1) {
                    out16[pos * 32 + n] = (_Float16)eluf(v);
                } else if constexpr (EPI == 2) {
                    out16[pos * 32 + n] =
                        (_Float16)eluf(v + (float)res[pos * 32 + n]);
                } else {
                    if (n < cout) out32[pos * 9 + n] = v;
                }
            }
        }
    }
}

// ---------------------------------------------------------------------------
// First conv: virtual 96-ch input [h1-col tile | h1-row tile | f2d | pad] -> 32
// Weight blocks = 3 K-chunks (kchunks=3, taps=1), NT=2.
// ---------------------------------------------------------------------------
__global__ __launch_bounds__(128) void conv_first(
    const float* __restrict__ f2d, const float* __restrict__ h1,
    const unsigned* __restrict__ wpk, const float* __restrict__ bias,
    _Float16* __restrict__ out16) {
    __shared__ float    h1l[16 * LL];
    __shared__ unsigned wlds[3 * 2 * 256];
    for (int i = threadIdx.x * 4; i < 16 * LL; i += 128 * 4)
        g2l_async_b128(&h1l[i], &h1[i]);
    for (int i = threadIdx.x * 4; i < 3 * 2 * 256; i += 128 * 4)
        g2l_async_b128(&wlds[i], &wpk[i]);
    wait_async();
    __syncthreads();

    int wave = threadIdx.x >> 5, lane = threadIdx.x & 31;
    int m = lane & 15, hi = lane >> 4;

    for (int ti = 0; ti < 4; ++ti) {
        int tile = (blockIdx.x * 4 + wave) * 4 + ti;
        int y = tile >> 5, x0 = (tile & 31) << 4;
        int x = x0 + m;

        v8f acc[2];
#pragma unroll
        for (int q = 0; q < 2; ++q)
#pragma unroll
            for (int r = 0; r < 8; ++r) acc[q][r] = 0.f;

        for (int kc = 0; kc < 3; ++kc) {
            AFrag A;
#pragma unroll
            for (int r = 0; r < 8; ++r) {
                int kb = kbase_of(r, hi);
                float a0, a1;
                if (kc == 0) {
                    a0 = (kb < 16) ? h1l[kb * LL + x] : h1l[(kb - 16) * LL + y];
                    int k1 = kb + 1;
                    a1 = (k1 < 16) ? h1l[k1 * LL + x] : h1l[(k1 - 16) * LL + y];
                } else {
                    int c0 = kc * 32 - 32 + kb;   // f2d channel
                    size_t sp = (size_t)y * LL + x;
                    a0 = (c0 < 41) ? f2d[(size_t)c0 * NPOS + sp] : 0.f;
                    a1 = (c0 + 1 < 41) ? f2d[(size_t)(c0 + 1) * NPOS + sp] : 0.f;
                }
                A.h[2 * r]     = (_Float16)a0;
                A.h[2 * r + 1] = (_Float16)a1;
            }
#pragma unroll
            for (int q = 0; q < 2; ++q) {
                AFrag B;
                const uint4* wq =
                    (const uint4*)&wlds[((kc * 2 + q) * 32 + lane) * 8];
                B.x[0] = wq[0];
                B.x[1] = wq[1];
                acc[q] = wmma16(A.v, B.v, acc[q]);
            }
        }
#pragma unroll
        for (int q = 0; q < 2; ++q) {
            int n = q * 16 + m;
            float bv = bias[n];
#pragma unroll
            for (int r = 0; r < 8; ++r) {
                int mm = hi * 8 + r;
                size_t pos = (size_t)y * LL + x0 + mm;
                out16[pos * 32 + n] = (_Float16)(acc[q][r] + bv);
            }
        }
    }
}

// ---------------------------------------------------------------------------
// Instance-norm stats (sum / sumsq per channel) and apply (+ELU / +res+ELU)
// ---------------------------------------------------------------------------
__global__ __launch_bounds__(256) void k_stats(const _Float16* __restrict__ x,
                                               float* __restrict__ acc) {
    __shared__ float ls[64];
    int t = threadIdx.x, c = t & 31, g = t >> 5;
    float s = 0.f, ss = 0.f;
    for (int p = blockIdx.x * 8 + g; p < NPOS; p += gridDim.x * 8) {
        float v = (float)x[(size_t)p * 32 + c];
        s += v;
        ss += v * v;
    }
    if (t < 64) ls[t] = 0.f;
    __syncthreads();
    atomicAdd(&ls[c], s);         // ds_add_f32
    atomicAdd(&ls[32 + c], ss);
    __syncthreads();
    if (t < 64) atomicAdd(&acc[t], ls[t]);   // global_atomic_add_f32
}

template <bool RES>
__global__ __launch_bounds__(256) void k_inorm_apply(
    const _Float16* __restrict__ in, const _Float16* __restrict__ res,
    _Float16* __restrict__ out, const float* __restrict__ g,
    const float* __restrict__ b, const float* __restrict__ acc) {
    __shared__ float sc[32], sh[32];
    if (threadIdx.x < 32) {
        int c = threadIdx.x;
        float mean = acc[c] * (1.f / NPOS);
        float var  = acc[32 + c] * (1.f / NPOS) - mean * mean;
        float s    = g[c] * rsqrtf(var + 1e-6f);
        sc[c] = s;
        sh[c] = b[c] - s * mean;
    }
    __syncthreads();
    size_t p = (size_t)blockIdx.x * blockDim.x + threadIdx.x;
    if (p >= NPOS) return;
    for (int c = 0; c < 32; ++c) {
        float v = (float)in[p * 32 + c] * sc[c] + sh[c];
        if constexpr (RES) v += (float)res[p * 32 + c];
        out[p * 32 + c] = (_Float16)eluf(v);
    }
}

// ---------------------------------------------------------------------------
// binp symmetrize  +  lddt
// ---------------------------------------------------------------------------
__global__ void k_sym(const float* __restrict__ raw, float* __restrict__ outb) {
    int idx = blockIdx.x * blockDim.x + threadIdx.x;
    if (idx >= NPOS) return;
    int i = idx >> 9, j = idx & 511;
    for (int c = 0; c < 9; ++c) {
        float v = 0.5f * (raw[(size_t)idx * 9 + c] +
                          raw[((size_t)j * LL + i) * 9 + c]);
        outb[(size_t)c * NPOS + idx] = v;
    }
}

__global__ __launch_bounds__(256) void k_lddt(const float* __restrict__ binp,
                                              const float* __restrict__ cmap,
                                              float* __restrict__ lddt) {
    int j = blockIdx.x;
    float p0 = 0, p1 = 0, p2 = 0, p3 = 0, p4 = 0;
    for (int i = threadIdx.x; i < LL; i += 256) {
        float vals[9], mx = -1e30f;
        for (int c = 0; c < 9; ++c) {
            vals[c] = binp[(size_t)c * NPOS + i * LL + j];
            mx = fmaxf(mx, vals[c]);
        }
        float s = 0.f;
        for (int c = 0; c < 9; ++c) {
            vals[c] = __expf(vals[c] - mx);
            s += vals[c];
        }
        float inv = 1.f / s;
        float msk = cmap[i * LL + j] * ((i == j) ? 0.f : 1.f);
        p0 += vals[4] * inv * msk;
        p1 += (vals[3] + vals[5]) * inv * msk;
        p2 += (vals[2] + vals[6]) * inv * msk;
        p3 += (vals[1] + vals[7]) * inv * msk;
        p4 += msk;
    }
    __shared__ float red[256];
    float vv[5] = {p0, p1, p2, p3, p4}, tot[5];
    for (int q = 0; q < 5; ++q) {
        red[threadIdx.x] = vv[q];
        __syncthreads();
        for (int o = 128; o > 0; o >>= 1) {
            if (threadIdx.x < o) red[threadIdx.x] += red[threadIdx.x + o];
            __syncthreads();
        }
        tot[q] = red[0];
        __syncthreads();
    }
    if (threadIdx.x == 0)
        lddt[j] = 0.25f * (4.f * tot[0] + 3.f * tot[1] + 2.f * tot[2] + tot[3]) /
                  tot[4];
}

// ---------------------------------------------------------------------------
// SE(3) attention pipeline
// ---------------------------------------------------------------------------
__global__ void k_nodeinit(const float* __restrict__ h1,
                           const float* __restrict__ lddt,
                           float* __restrict__ h) {
    int l = blockIdx.x * blockDim.x + threadIdx.x;
    if (l >= LL) return;
    for (int c = 0; c < 16; ++c) h[l * 17 + c] = h1[c * LL + l];
    h[l * 17 + 16] = lddt[l];
}

__global__ void k_we(const _Float16* __restrict__ base,
                     const float* __restrict__ binp, const int* __restrict__ el,
                     float* __restrict__ we) {
    int e = blockIdx.x * blockDim.x + threadIdx.x;
    if (e >= EDGES) return;
    int s = el[e], d = el[EDGES + e];
    size_t pp = (size_t)s * LL + d;
    for (int c = 0; c < 32; ++c) we[(size_t)e * 41 + c] = (float)base[pp * 32 + c];
    for (int c = 0; c < 9; ++c)
        we[(size_t)e * 41 + 32 + c] = binp[(size_t)c * NPOS + pp];
}

__global__ void k_q(const float* __restrict__ h, const float* __restrict__ Wq,
                    float* __restrict__ q, int cin) {
    int idx = blockIdx.x * blockDim.x + threadIdx.x;
    if (idx >= LL * 16) return;
    int l = idx >> 4, hh = idx & 15;
    float s = 0.f;
    for (int c = 0; c < cin; ++c) s += h[l * 17 + c] * Wq[c * 16 + hh];
    q[idx] = s;
}

template <int CIN, int COUT>
__global__ __launch_bounds__(256) void k_edge(
    const float* __restrict__ h, const float* __restrict__ we,
    const float* __restrict__ q, const int* __restrict__ el,
    const float* __restrict__ Wk, const float* __restrict__ Wv0,
    const float* __restrict__ Wv1, float* __restrict__ logits,
    float* __restrict__ v0, float* __restrict__ v1) {
    constexpr int SF = CIN + 41;
    __shared__ float wk[SF * 16];
    __shared__ float wv0[SF * COUT];
    __shared__ float wv1[SF];
    for (int i = threadIdx.x; i < SF * 16; i += 256) wk[i] = Wk[i];
    for (int i = threadIdx.x; i < SF * COUT; i += 256) wv0[i] = Wv0[i];
    for (int i = threadIdx.x; i < SF; i += 256) wv1[i] = Wv1[i];
    __syncthreads();
    int e = blockIdx.x * 256 + threadIdx.x;
    if (e >= EDGES) return;
    int s = el[e], d = el[EDGES + e];
    float sf[SF];
    for (int c = 0; c < CIN; ++c) sf[c] = h[s * 17 + c];
    for (int c = 0; c < 41; ++c) sf[CIN + c] = we[(size_t)e * 41 + c];
    float lg = 0.f;
    for (int hh = 0; hh < 16; ++hh) {
        float kk = 0.f;
        for (int c = 0; c < SF; ++c) kk += sf[c] * wk[c * 16 + hh];
        lg += q[d * 16 + hh] * kk;
    }
    logits[e] = lg * 0.25f;   // 1/sqrt(16)
    for (int o = 0; o < COUT; ++o) {
        float v = 0.f;
        for (int c = 0; c < SF; ++c) v += sf[c] * wv0[c * COUT + o];
        v0[(size_t)e * COUT + o] = v;
    }
    float vv = 0.f;
    for (int c = 0; c < SF; ++c) vv += sf[c] * wv1[c];
    v1[e] = vv;
}

template <int CIN, int COUT, bool DOELU>
__global__ __launch_bounds__(128) void k_nodered(
    const float* __restrict__ h, const float* __restrict__ logits,
    const float* __restrict__ v0, const float* __restrict__ v1,
    const int* __restrict__ el, const float* __restrict__ pos,
    const float* __restrict__ Wskip, float* __restrict__ hout,
    float* __restrict__ xacc) {
    int n = blockIdx.x;
    __shared__ float red[128];
    // pass 1: segment max
    float mx = -1e30f;
    for (int e = threadIdx.x; e < EDGES; e += 128)
        if (el[EDGES + e] == n) mx = fmaxf(mx, logits[e]);
    red[threadIdx.x] = mx;
    __syncthreads();
    for (int o = 64; o > 0; o >>= 1) {
        if (threadIdx.x < o)
            red[threadIdx.x] = fmaxf(red[threadIdx.x], red[threadIdx.x + o]);
        __syncthreads();
    }
    mx = red[0];
    __syncthreads();
    // pass 2: sum of exp
    float sm = 0.f;
    for (int e = threadIdx.x; e < EDGES; e += 128)
        if (el[EDGES + e] == n) sm += __expf(logits[e] - mx);
    red[threadIdx.x] = sm;
    __syncthreads();
    for (int o = 64; o > 0; o >>= 1) {
        if (threadIdx.x < o) red[threadIdx.x] += red[threadIdx.x + o];
        __syncthreads();
    }
    float ssum = red[0] + 1e-9f;
    __syncthreads();
    // pass 3: weighted sums
    float part[COUT + 3];
    for (int q = 0; q < COUT + 3; ++q) part[q] = 0.f;
    float pn0 = pos[n * 3], pn1 = pos[n * 3 + 1], pn2 = pos[n * 3 + 2];
    for (int e = threadIdx.x; e < EDGES; e += 128) {
        if (el[EDGES + e] != n) continue;
        float a = __expf(logits[e] - mx) / ssum;
        for (int o = 0; o < COUT; ++o) part[o] += a * v0[(size_t)e * COUT + o];
        int s = el[e];
        float av1 = a * v1[e];
        part[COUT]     += av1 * (pos[s * 3]     - pn0);
        part[COUT + 1] += av1 * (pos[s * 3 + 1] - pn1);
        part[COUT + 2] += av1 * (pos[s * 3 + 2] - pn2);
    }
    __shared__ float tot[COUT + 3];
    for (int q = 0; q < COUT + 3; ++q) {
        red[threadIdx.x] = part[q];
        __syncthreads();
        for (int o = 64; o > 0; o >>= 1) {
            if (threadIdx.x < o) red[threadIdx.x] += red[threadIdx.x + o];
            __syncthreads();
        }
        if (threadIdx.x == 0) tot[q] = red[0];
        __syncthreads();
    }
    if (threadIdx.x < COUT) {
        int c = threadIdx.x;
        float v = tot[c];
        for (int ci = 0; ci < CIN; ++ci) v += h[n * 17 + ci] * Wskip[ci * COUT + c];
        hout[n * 17 + c] = DOELU ? eluf(v) : v;
    }
    if (threadIdx.x < 3) xacc[n * 3 + threadIdx.x] += tot[COUT + threadIdx.x];
}

__global__ void k_final(const float* __restrict__ pos,
                        const float* __restrict__ xacc,
                        const float* __restrict__ lddt,
                        const float* __restrict__ h, float* __restrict__ out) {
    int l = blockIdx.x * blockDim.x + threadIdx.x;
    if (l >= LL) return;
    for (int d = 0; d < 3; ++d)
        out[(size_t)9 * NPOS + l * 3 + d] = pos[l * 3 + d] + xacc[l * 3 + d];
    out[(size_t)9 * NPOS + 1536 + l] = lddt[l] + tanhf(h[l * 17]);
}

// ---------------------------------------------------------------------------
// Workspace layout
// ---------------------------------------------------------------------------
static constexpr size_t alup(size_t x) { return (x + 255) & ~(size_t)255; }
static constexpr size_t SZ_ACT = (size_t)NPOS * 32 * 2;            // 16 MiB
static constexpr size_t O_BUF1 = 0;
static constexpr size_t O_BUF2 = O_BUF1 + SZ_ACT;
static constexpr size_t O_BUF3 = O_BUF2 + SZ_ACT;
static constexpr size_t O_BASE = O_BUF3 + SZ_ACT;
static constexpr size_t O_RAW9 = O_BASE + SZ_ACT;
static constexpr size_t O_WPK  = O_RAW9 + alup((size_t)NPOS * 9 * 4);
static constexpr size_t O_ACC  = O_WPK + alup(4608 * 4);
static constexpr size_t O_H1   = O_ACC + 256;
static constexpr size_t O_LDDT = O_H1 + alup(16 * LL * 4);
static constexpr size_t O_WE   = O_LDDT + alup(LL * 4);
static constexpr size_t O_Q    = O_WE + alup((size_t)EDGES * 41 * 4);
static constexpr size_t O_LOG  = O_Q + alup(LL * 16 * 4);
static constexpr size_t O_V0   = O_LOG + alup(EDGES * 4);
static constexpr size_t O_V1   = O_V0 + alup((size_t)EDGES * 16 * 4);
static constexpr size_t O_HA   = O_V1 + alup(EDGES * 4);
static constexpr size_t O_HB   = O_HA + alup(LL * 17 * 4);
static constexpr size_t O_XACC = O_HB + alup(LL * 17 * 4);

// ---------------------------------------------------------------------------
// Launch
// ---------------------------------------------------------------------------
extern "C" void kernel_launch(void* const* d_in, const int* in_sizes, int n_in,
                              void* d_out, int out_size, void* d_ws,
                              size_t ws_size, hipStream_t stream) {
    (void)in_sizes; (void)n_in; (void)out_size; (void)ws_size;
    const float* f1d  = (const float*)d_in[0];
    const float* f2d  = (const float*)d_in[1];
    const float* pos  = (const float*)d_in[2];
    const int*   el   = (const int*)d_in[3];
    const float* cmap = (const float*)d_in[4];
    auto P = [&](int i) { return (const float*)d_in[i]; };
    // param flattening (sorted pytree keys):
    // 5..44  base_blocks[i]: {b1,b2,be1,be2,g1,g2,w1,w2}
    // 45 base_proj_b  46 base_proj_w
    // 47..58 bin_blocks[i]: {b1,b2,w1,w2}
    // 59 bin_proj_b  60 bin_proj_w
    // 61 c1_b 62 c1_w 63 c2_b 64 c2_w 65 cbin_b 66 cbin_w 67 in1_b 68 in1_g
    // 69..83 se3[i]: {Wk,Wq,Wskip,Wv0,Wv1}
    auto SE = [&](int li, int j) { return (const float*)d_in[69 + 5 * li + j]; };

    char*      ws    = (char*)d_ws;
    _Float16*  buf1  = (_Float16*)(ws + O_BUF1);
    _Float16*  buf2  = (_Float16*)(ws + O_BUF2);
    _Float16*  buf3  = (_Float16*)(ws + O_BUF3);
    _Float16*  bbase = (_Float16*)(ws + O_BASE);
    float*     raw9  = (float*)(ws + O_RAW9);
    unsigned*  wpk   = (unsigned*)(ws + O_WPK);
    float*     accb  = (float*)(ws + O_ACC);
    float*     h1b   = (float*)(ws + O_H1);
    float*     lddtb = (float*)(ws + O_LDDT);
    float*     web   = (float*)(ws + O_WE);
    float*     qb    = (float*)(ws + O_Q);
    float*     logb  = (float*)(ws + O_LOG);
    float*     v0b   = (float*)(ws + O_V0);
    float*     v1b   = (float*)(ws + O_V1);
    float*     hA    = (float*)(ws + O_HA);
    float*     hB    = (float*)(ws + O_HB);
    float*     xaccb = (float*)(ws + O_XACC);
    float*     out   = (float*)d_out;

    // 1) h1 = elu(c1_w @ f1d)
    k_h1<<<8, 64, 0, stream>>>(f1d, P(62), P(61), h1b);

    // 2) first 1x1 conv (73->32) on virtual tiled input
    k_prep<<<8, 256, 0, stream>>>(P(64), wpk, 32, 73, 1, 3, 2);
    conv_first<<<1024, 128, 0, stream>>>(f2d, h1b, wpk, P(63), buf1);

    // 3) in1 instance norm + ELU
    k_zero<<<1, 64, 0, stream>>>(accb, 64);
    k_stats<<<256, 256, 0, stream>>>(buf1, accb);
    k_inorm_apply<false><<<1024, 256, 0, stream>>>(buf1, nullptr, buf2, P(68),
                                                   P(67), accb);

    // 4) base_proj 1x1
    k_prep<<<8, 256, 0, stream>>>(P(46), wpk, 32, 32, 1, 1, 2);
    conv_wmma<0, 2><<<1024, 128, 0, stream>>>(buf2, wpk, P(45), buf1, nullptr,
                                              nullptr, 1, 1, 32);

    // 5) 5 inorm res-blocks, dilations 1,2,4,8,1
    const int dils[5] = {1, 2, 4, 8, 1};
    for (int bi = 0; bi < 5; ++bi) {
        int b = 5 + 8 * bi;
        k_prep<<<8, 256, 0, stream>>>(P(b + 6), wpk, 32, 32, 9, 1, 2);
        conv_wmma<0, 2><<<1024, 128, 0, stream>>>(buf1, wpk, P(b + 0), buf2,
                                                  nullptr, nullptr, 9, dils[bi], 32);
        k_zero<<<1, 64, 0, stream>>>(accb, 64);
        k_stats<<<256, 256, 0, stream>>>(buf2, accb);
        k_inorm_apply<false><<<1024, 256, 0, stream>>>(buf2, nullptr, buf3,
                                                       P(b + 4), P(b + 2), accb);
        k_prep<<<8, 256, 0, stream>>>(P(b + 7), wpk, 32, 32, 9, 1, 2);
        conv_wmma<0, 2><<<1024, 128, 0, stream>>>(buf3, wpk, P(b + 1), buf2,
                                                  nullptr, nullptr, 9, dils[bi], 32);
        k_zero<<<1, 64, 0, stream>>>(accb, 64);
        k_stats<<<256, 256, 0, stream>>>(buf2, accb);
        k_inorm_apply<true><<<1024, 256, 0, stream>>>(buf2, buf1, buf1,
                                                      P(b + 5), P(b + 3), accb);
    }

    // 6) base = elu(x)
    k_elu16<<<4096, 256, 0, stream>>>(buf1, bbase, (size_t)NPOS * 32);

    // 7) bin path
    k_prep<<<8, 256, 0, stream>>>(P(60), wpk, 32, 32, 1, 1, 2);
    conv_wmma<0, 2><<<1024, 128, 0, stream>>>(bbase, wpk, P(59), buf1, nullptr,
                                              nullptr, 1, 1, 32);
    const int dils2[3] = {1, 2, 1};
    for (int bi = 0; bi < 3; ++bi) {
        int b = 47 + 4 * bi;
        k_prep<<<8, 256, 0, stream>>>(P(b + 2), wpk, 32, 32, 9, 1, 2);
        conv_wmma<1, 2><<<1024, 128, 0, stream>>>(buf1, wpk, P(b + 0), buf2,
                                                  nullptr, nullptr, 9, dils2[bi], 32);
        k_prep<<<8, 256, 0, stream>>>(P(b + 3), wpk, 32, 32, 9, 1, 2);
        conv_wmma<2, 2><<<1024, 128, 0, stream>>>(buf2, wpk, P(b + 1), buf1,
                                                  nullptr, buf1, 9, dils2[bi], 32);
    }
    k_elu16<<<4096, 256, 0, stream>>>(buf1, buf2, (size_t)NPOS * 32);

    // 8) cbin 1x1 -> 9 bins (f32), symmetrize into d_out, lddt
    k_prep<<<8, 256, 0, stream>>>(P(66), wpk, 9, 32, 1, 1, 1);
    conv_wmma<3, 1><<<1024, 128, 0, stream>>>(buf2, wpk, P(65), nullptr, raw9,
                                              nullptr, 1, 1, 9);
    k_sym<<<1024, 256, 0, stream>>>(raw9, out);
    k_lddt<<<512, 256, 0, stream>>>(out, cmap, lddtb);

    // 9) SE(3) attention
    k_nodeinit<<<2, 256, 0, stream>>>(h1b, lddtb, hA);
    k_we<<<64, 256, 0, stream>>>(bbase, out, el, web);
    k_zero<<<6, 256, 0, stream>>>(xaccb, LL * 3);

    // layer 0: 17 -> 16
    k_q<<<32, 256, 0, stream>>>(hA, SE(0, 1), qb, 17);
    k_edge<17, 16><<<64, 256, 0, stream>>>(hA, web, qb, el, SE(0, 0), SE(0, 3),
                                           SE(0, 4), logb, v0b, v1b);
    k_nodered<17, 16, true><<<512, 128, 0, stream>>>(hA, logb, v0b, v1b, el, pos,
                                                     SE(0, 2), hB, xaccb);
    // layer 1: 16 -> 16
    k_q<<<32, 256, 0, stream>>>(hB, SE(1, 1), qb, 16);
    k_edge<16, 16><<<64, 256, 0, stream>>>(hB, web, qb, el, SE(1, 0), SE(1, 3),
                                           SE(1, 4), logb, v0b, v1b);
    k_nodered<16, 16, true><<<512, 128, 0, stream>>>(hB, logb, v0b, v1b, el, pos,
                                                     SE(1, 2), hA, xaccb);
    // layer 2: 16 -> 1 (no elu)
    k_q<<<32, 256, 0, stream>>>(hA, SE(2, 1), qb, 16);
    k_edge<16, 1><<<64, 256, 0, stream>>>(hA, web, qb, el, SE(2, 0), SE(2, 3),
                                          SE(2, 4), logb, v0b, v1b);
    k_nodered<16, 1, false><<<512, 128, 0, stream>>>(hA, logb, v0b, v1b, el, pos,
                                                     SE(2, 2), hB, xaccb);

    // 10) outputs: x_new, lddt_final
    k_final<<<2, 256, 0, stream>>>(pos, xaccb, lddtb, hB, out);
}
